// FeedbackModelUDA_74311524155810
// MI455X (gfx1250) — compile-verified, matrix operand order checked by reading, not run on hardware
//
#include <hip/hip_runtime.h>
#include <hip/hip_bf16.h>

// ---------------------------------------------------------------------------
// FeedbackModel (biLSTM -> span pool -> MHA -> classifier) for MI455X gfx1250
// All heavy GEMMs use v_wmma_f32_16x16x32_bf16 (bf16 A/B, f32 accumulate).
// ---------------------------------------------------------------------------

#define B_   16
#define S_   2048
#define H_   1024
#define HC_  512
#define G4_  2048   // 4*HC
#define NS_  16
#define NH_  16
#define DH_  64
#define NL_  3
#define EPS_ 1e-7f

typedef __attribute__((ext_vector_type(16))) __bf16 v16bf;
typedef __attribute__((ext_vector_type(8)))  __bf16 v8bf;
typedef __attribute__((ext_vector_type(8)))  float  v8f;

// CDNA5 16-bit A/B fragment load (row-major [row][k] matrix, 16 bf16 per lane).
// A layout (ISA 7.12.2): lanes 0-15 hold row M=lane, K = {k0..k0+7, k0+16..k0+23};
// lanes 16-31 hold row M=lane-16, K = {k0+8..k0+15, k0+24..k0+31}.
// B mirrors this with lane&15 selecting the N column; weights are stored
// transposed ([n][k]) so K is contiguous for both operands -> b128 loads.
__device__ __forceinline__ v16bf ld_frag(const __bf16* rowk, int lane) {
    const int hi = (lane >> 4) & 1;
    v8bf lo = *reinterpret_cast<const v8bf*>(rowk + hi * 8);
    v8bf hv = *reinterpret_cast<const v8bf*>(rowk + 16 + hi * 8);
    v16bf r;
#pragma unroll
    for (int j = 0; j < 8; ++j) { r[j] = lo[j]; r[j + 8] = hv[j]; }
    return r;
}

__device__ __forceinline__ float sigmoidf_(float x) { return 1.0f / (1.0f + expf(-x)); }

// -------------------------- conversion kernels -----------------------------
__global__ void k_cvt_bf16(const float* __restrict__ src, __bf16* __restrict__ dst, size_t n) {
    size_t i = (size_t)blockIdx.x * blockDim.x + threadIdx.x;
    size_t stride = (size_t)gridDim.x * blockDim.x;
    for (; i < n; i += stride) dst[i] = (__bf16)src[i];
}

// src [R][C] f32 -> dst [C][R] bf16 (transpose-convert, for x@W style weights)
__global__ void k_cvtT_bf16(const float* __restrict__ src, __bf16* __restrict__ dst, int R, int C) {
    size_t n = (size_t)R * C;
    size_t i = (size_t)blockIdx.x * blockDim.x + threadIdx.x;
    size_t stride = (size_t)gridDim.x * blockDim.x;
    for (; i < n; i += stride) {
        int ci = (int)(i / R);      // output row (= src column)
        int ri = (int)(i - (size_t)ci * R);
        dst[i] = (__bf16)src[(size_t)ri * C + ci];
    }
}

// ------------------- LSTM input projection (WMMA GEMM) ---------------------
// xp[dir][b*S+t][g] = x[b,t,:] . Wih[dir][g,:] + bih[g] + bhh[g]
// Each wave owns 4 M-tiles x 1 N-tile: one B-frag load feeds 4 WMMAs.
// Software-pipelined over K: fragments of chunk k+1 are loaded (rotating
// register sets) before the WMMAs of chunk k are issued, so the scheduler
// cannot sink loads to immediately before their consumers and waits retire
// with loads still in flight.
#define MT_ 4
__global__ __launch_bounds__(256) void k_input_proj(
    const __bf16* __restrict__ xb,     // [32768][1024]
    const __bf16* __restrict__ wih,    // [2][2048][1024]  (row g contiguous in k)
    const float* __restrict__ bih_f, const float* __restrict__ bhh_f,
    const float* __restrict__ bih_b, const float* __restrict__ bhh_b,
    float* __restrict__ xp)            // [2][32768][2048]
{
    const int wave = threadIdx.x >> 5, lane = threadIdx.x & 31;
    const int m0 = blockIdx.x * (16 * MT_);
    const int n0 = (blockIdx.y * 8 + wave) * 16;
    const int dir = blockIdx.z;
    const int n = n0 + (lane & 15);
    const __bf16* arow = xb + (size_t)(m0 + (lane & 15)) * H_;
    const __bf16* brow = wih + (size_t)dir * G4_ * H_ + (size_t)n * H_;
    v8f acc[MT_];
#pragma unroll
    for (int m = 0; m < MT_; ++m) acc[m] = (v8f){};

    // prologue: load chunk 0
    v16bf bcur = ld_frag(brow, lane);
    v16bf acur[MT_];
#pragma unroll
    for (int m = 0; m < MT_; ++m) acur[m] = ld_frag(arow + (size_t)m * 16 * H_, lane);

    for (int k0 = 32; k0 < H_; k0 += 32) {
        // issue next chunk's loads before consuming the current chunk
        v16bf bnxt = ld_frag(brow + k0, lane);
        v16bf anxt[MT_];
#pragma unroll
        for (int m = 0; m < MT_; ++m) anxt[m] = ld_frag(arow + (size_t)m * 16 * H_ + k0, lane);
#pragma unroll
        for (int m = 0; m < MT_; ++m)
            acc[m] = __builtin_amdgcn_wmma_f32_16x16x32_bf16(false, acur[m], false, bcur,
                                                             (short)0, acc[m], false, false);
        bcur = bnxt;
#pragma unroll
        for (int m = 0; m < MT_; ++m) acur[m] = anxt[m];
    }
    // epilogue: last chunk
#pragma unroll
    for (int m = 0; m < MT_; ++m)
        acc[m] = __builtin_amdgcn_wmma_f32_16x16x32_bf16(false, acur[m], false, bcur,
                                                         (short)0, acc[m], false, false);

    const float bias = (dir ? (bih_b[n] + bhh_b[n]) : (bih_f[n] + bhh_f[n]));
    const int rbase = (lane < 16) ? 0 : 8;
    float* out = xp + (size_t)dir * (size_t)B_ * S_ * G4_;
#pragma unroll
    for (int m = 0; m < MT_; ++m)
#pragma unroll
        for (int r = 0; r < 8; ++r)
            out[(size_t)(m0 + m * 16 + rbase + r) * G4_ + n] = acc[m][r] + bias;
}

// ------------------------- LSTM recurrence (WMMA) --------------------------
// One workgroup per direction; h double-buffered in LDS (bf16), c in VGPRs.
// 32 waves x (4 gate tiles) x (16 K-chunks) = 64 WMMAs per step per wave.
// xp loads are hoisted off the critical path: GEMM accumulates from zero,
// xp is fetched into independent registers before the K-loop (overlaps with
// the WMMA chain) and added during the gate nonlinearity. The K-loop is
// software-pipelined (rotating register sets) so the four Whh B-frag loads
// of chunk k+1 overlap the WMMAs of chunk k.
__global__ __launch_bounds__(1024) void k_lstm_scan(
    const float* __restrict__ xp,     // [2][B*S][2048]
    const __bf16* __restrict__ whh,   // [2][2048][512]
    float* __restrict__ h_hist)       // [B][S][1024]  (dir0 -> cols 0..511, dir1 -> 512..1023)
{
    __shared__ __bf16 hbuf[2][B_ * HC_];   // 2 x 16KB double buffer
    const int dir = blockIdx.x;
    const int tid = threadIdx.x;
    const int wave = tid >> 5, lane = tid & 31;
    const int col = wave * 16 + (lane & 15);   // hidden column owned by this lane
    const int rbase = (lane < 16) ? 0 : 8;

    for (int i = tid; i < 2 * B_ * HC_; i += 1024) (&hbuf[0][0])[i] = (__bf16)0.0f;
    __syncthreads();

    const __bf16* W = whh + (size_t)dir * G4_ * HC_;
    const float* xpd = xp + (size_t)dir * (size_t)B_ * S_ * G4_;

    float cst[8];
#pragma unroll
    for (int r = 0; r < 8; ++r) cst[r] = 0.0f;

    for (int step = 0; step < S_; ++step) {
        const int t = dir ? (S_ - 1 - step) : step;
        const __bf16* hread = hbuf[step & 1];
        __bf16* hwrite = hbuf[(step + 1) & 1];

        // issue xp loads early (independent of the WMMA chain)
        float xv[4][8];
#pragma unroll
        for (int q = 0; q < 4; ++q) {
            const int gc = q * HC_ + col;
#pragma unroll
            for (int r = 0; r < 8; ++r)
                xv[q][r] = xpd[((size_t)(rbase + r) * S_ + t) * G4_ + gc];
        }
        // prefetch next step's xp rows into cache while we compute this step
        if (step + 1 < S_) {
            const int tn = dir ? (S_ - 2 - step) : (step + 1);
#pragma unroll
            for (int q = 0; q < 4; ++q)
                __builtin_prefetch(&xpd[((size_t)rbase * S_ + tn) * G4_ + q * HC_ + col], 0, 3);
        }

        v8f acc[4];
#pragma unroll
        for (int q = 0; q < 4; ++q) acc[q] = (v8f){};

        const __bf16* arow = hread + (size_t)(lane & 15) * HC_;

        // prologue: chunk 0
        v16bf acur = ld_frag(arow, lane);          // ds_load from LDS
        v16bf bcur[4];
#pragma unroll
        for (int q = 0; q < 4; ++q)
            bcur[q] = ld_frag(W + (size_t)(q * HC_ + col) * HC_, lane);

        for (int k0 = 32; k0 < HC_; k0 += 32) {
            v16bf anxt = ld_frag(arow + k0, lane);
            v16bf bnxt[4];
#pragma unroll
            for (int q = 0; q < 4; ++q)
                bnxt[q] = ld_frag(W + (size_t)(q * HC_ + col) * HC_ + k0, lane);
#pragma unroll
            for (int q = 0; q < 4; ++q)
                acc[q] = __builtin_amdgcn_wmma_f32_16x16x32_bf16(false, acur, false, bcur[q],
                                                                 (short)0, acc[q], false, false);
            acur = anxt;
#pragma unroll
            for (int q = 0; q < 4; ++q) bcur[q] = bnxt[q];
        }
        // epilogue: last chunk
#pragma unroll
        for (int q = 0; q < 4; ++q)
            acc[q] = __builtin_amdgcn_wmma_f32_16x16x32_bf16(false, acur, false, bcur[q],
                                                             (short)0, acc[q], false, false);

        // gates: i,f,g,o (torch order)
#pragma unroll
        for (int r = 0; r < 8; ++r) {
            const float ig = sigmoidf_(acc[0][r] + xv[0][r]);
            const float fg = sigmoidf_(acc[1][r] + xv[1][r]);
            const float gg = tanhf(acc[2][r] + xv[2][r]);
            const float og = sigmoidf_(acc[3][r] + xv[3][r]);
            const float c = fg * cst[r] + ig * gg;
            cst[r] = c;
            const float h = og * tanhf(c);
            const int b = rbase + r;
            hwrite[b * HC_ + col] = (__bf16)h;
            h_hist[((size_t)b * S_ + t) * H_ + dir * HC_ + col] = h;
        }
        __syncthreads();
    }
}

// --------------------- span mean pooling + layer norm ----------------------
__global__ __launch_bounds__(256) void k_span_pool(
    const float* __restrict__ h_hist, const int* __restrict__ heads,
    const int* __restrict__ tails, const float* __restrict__ ln_g,
    const float* __restrict__ ln_b, float* __restrict__ pooled,
    __bf16* __restrict__ pooled16)
{
    const int b = blockIdx.x, k = blockIdx.y;
    const int t0 = heads[b * NS_ + k] + 1;
    const int t1 = tails[b * NS_ + k];
    const float inv = 1.0f / (float)max(t1 - t0, 1);
    const int tid = threadIdx.x;
    float v[4] = {0, 0, 0, 0};
    for (int t = t0; t < t1; ++t) {
        const float* row = h_hist + ((size_t)b * S_ + t) * H_ + tid * 4;
#pragma unroll
        for (int j = 0; j < 4; ++j) v[j] += row[j];
    }
#pragma unroll
    for (int j = 0; j < 4; ++j) v[j] *= inv;

    __shared__ float s1[256], s2[256];
    float a1 = 0, a2 = 0;
#pragma unroll
    for (int j = 0; j < 4; ++j) { a1 += v[j]; a2 += v[j] * v[j]; }
    s1[tid] = a1; s2[tid] = a2;
    __syncthreads();
    for (int s = 128; s > 0; s >>= 1) {
        if (tid < s) { s1[tid] += s1[tid + s]; s2[tid] += s2[tid + s]; }
        __syncthreads();
    }
    const float mean = s1[0] * (1.0f / H_);
    const float var = s2[0] * (1.0f / H_) - mean * mean;
    const float rs = rsqrtf(var + EPS_);
    float* po = pooled + ((size_t)b * NS_ + k) * H_ + tid * 4;
    __bf16* po16 = pooled16 + ((size_t)b * NS_ + k) * H_ + tid * 4;
#pragma unroll
    for (int j = 0; j < 4; ++j) {
        const int f = tid * 4 + j;
        const float o = (v[j] - mean) * rs * ln_g[f] + ln_b[f];
        po[j] = o; po16[j] = (__bf16)o;
    }
}

// ------------------------ Q/K/V projections (WMMA) -------------------------
__global__ __launch_bounds__(256) void k_qkv(
    const __bf16* __restrict__ pooled16,   // [B][16][1024]
    const __bf16* __restrict__ wqkvT,      // [3][1024][1024] (n-major, k contiguous)
    const float* __restrict__ bq, const float* __restrict__ bk,
    const float* __restrict__ bv, float* __restrict__ qkv)  // [3][B][16][1024]
{
    const int b = blockIdx.x, z = blockIdx.z;
    const int wave = threadIdx.x >> 5, lane = threadIdx.x & 31;
    const int n = (blockIdx.y * 8 + wave) * 16 + (lane & 15);
    const __bf16* arow = pooled16 + (size_t)b * NS_ * H_ + (size_t)(lane & 15) * H_;
    const __bf16* brow = wqkvT + (size_t)z * H_ * H_ + (size_t)n * H_;
    v8f acc = {};
#pragma unroll 4
    for (int k0 = 0; k0 < H_; k0 += 32)
        acc = __builtin_amdgcn_wmma_f32_16x16x32_bf16(false, ld_frag(arow + k0, lane),
                                                      false, ld_frag(brow + k0, lane),
                                                      (short)0, acc, false, false);
    const float* bias = (z == 0) ? bq : (z == 1 ? bk : bv);
    const float bb = bias[n];
    const int rbase = (lane < 16) ? 0 : 8;
    float* out = qkv + ((size_t)z * B_ + b) * NS_ * H_;
#pragma unroll
    for (int r = 0; r < 8; ++r) out[(size_t)(rbase + r) * H_ + n] = acc[r] + bb;
}

// -------------------- attention core (tiny, scalar math) -------------------
__global__ __launch_bounds__(256) void k_attn(
    const float* __restrict__ qkv, const int* __restrict__ amask,
    float* __restrict__ ctx, __bf16* __restrict__ ctx16)
{
    const int b = blockIdx.x, h = blockIdx.y;
    const int tid = threadIdx.x;
    const int qi = tid >> 4, ki = tid & 15;
    const float* q = qkv + ((size_t)0 * B_ + b) * NS_ * H_;
    const float* k = qkv + ((size_t)1 * B_ + b) * NS_ * H_;
    const float* v = qkv + ((size_t)2 * B_ + b) * NS_ * H_;
    __shared__ float sc[NS_][NS_], pr[NS_][NS_];
    const float* qr = q + (size_t)qi * H_ + h * DH_;
    const float* kr = k + (size_t)ki * H_ + h * DH_;
    float s = 0.0f;
    for (int d = 0; d < DH_; ++d) s += qr[d] * kr[d];
    s *= 0.125f;  // 1/sqrt(64)
    if (!(amask[b * NS_ + qi] > 0 && amask[b * NS_ + ki] > 0)) s = -3.402823466e38f;
    sc[qi][ki] = s;
    __syncthreads();
    if (tid < NS_) {
        float mx = sc[tid][0];
        for (int j = 1; j < NS_; ++j) mx = fmaxf(mx, sc[tid][j]);
        float sum = 0.0f;
        for (int j = 0; j < NS_; ++j) { float e = expf(sc[tid][j] - mx); pr[tid][j] = e; sum += e; }
        const float inv = 1.0f / sum;
        for (int j = 0; j < NS_; ++j) pr[tid][j] *= inv;
    }
    __syncthreads();
    const int d0 = (tid & 15) * 4;
    float acc[4] = {0, 0, 0, 0};
    for (int kk = 0; kk < NS_; ++kk) {
        const float p = pr[qi][kk];
        const float* vr = v + (size_t)kk * H_ + h * DH_ + d0;
#pragma unroll
        for (int j = 0; j < 4; ++j) acc[j] += p * vr[j];
    }
    float* co = ctx + ((size_t)b * NS_ + qi) * H_ + h * DH_ + d0;
    __bf16* co16 = ctx16 + ((size_t)b * NS_ + qi) * H_ + h * DH_ + d0;
#pragma unroll
    for (int j = 0; j < 4; ++j) { co[j] = acc[j]; co16[j] = (__bf16)acc[j]; }
}

// ------------------------ output projection (WMMA) -------------------------
__global__ __launch_bounds__(256) void k_oproj(
    const __bf16* __restrict__ ctx16, const __bf16* __restrict__ woT,
    const float* __restrict__ bo, float* __restrict__ attn)
{
    const int b = blockIdx.x;
    const int wave = threadIdx.x >> 5, lane = threadIdx.x & 31;
    const int n = (blockIdx.y * 8 + wave) * 16 + (lane & 15);
    const __bf16* arow = ctx16 + (size_t)b * NS_ * H_ + (size_t)(lane & 15) * H_;
    const __bf16* brow = woT + (size_t)n * H_;
    v8f acc = {};
#pragma unroll 4
    for (int k0 = 0; k0 < H_; k0 += 32)
        acc = __builtin_amdgcn_wmma_f32_16x16x32_bf16(false, ld_frag(arow + k0, lane),
                                                      false, ld_frag(brow + k0, lane),
                                                      (short)0, acc, false, false);
    const float bb = bo[n];
    const int rbase = (lane < 16) ? 0 : 8;
    float* out = attn + (size_t)b * NS_ * H_;
#pragma unroll
    for (int r = 0; r < 8; ++r) out[(size_t)(rbase + r) * H_ + n] = acc[r] + bb;
}

// ------------------ residual + layer norm + classifier ---------------------
__global__ __launch_bounds__(256) void k_final(
    const float* __restrict__ attn, const float* __restrict__ pooled,
    const float* __restrict__ aln_g, const float* __restrict__ aln_b,
    const float* __restrict__ Wc, const float* __restrict__ bc,
    float* __restrict__ out)   // [B][NS][3]
{
    const int b = blockIdx.x, k = blockIdx.y;
    const int tid = threadIdx.x;
    const size_t base = ((size_t)b * NS_ + k) * H_ + tid * 4;
    float v[4];
#pragma unroll
    for (int j = 0; j < 4; ++j) v[j] = attn[base + j] + pooled[base + j];

    __shared__ float s1[256], s2[256];
    float a1 = 0, a2 = 0;
#pragma unroll
    for (int j = 0; j < 4; ++j) { a1 += v[j]; a2 += v[j] * v[j]; }
    s1[tid] = a1; s2[tid] = a2;
    __syncthreads();
    for (int s = 128; s > 0; s >>= 1) {
        if (tid < s) { s1[tid] += s1[tid + s]; s2[tid] += s2[tid + s]; }
        __syncthreads();
    }
    const float mean = s1[0] * (1.0f / H_);
    const float var = s2[0] * (1.0f / H_) - mean * mean;
    const float rs = rsqrtf(var + EPS_);
    float part[NL_] = {0, 0, 0};
#pragma unroll
    for (int j = 0; j < 4; ++j) {
        const int f = tid * 4 + j;
        const float fe = (v[j] - mean) * rs * aln_g[f] + aln_b[f];
#pragma unroll
        for (int c = 0; c < NL_; ++c) part[c] += fe * Wc[(size_t)f * NL_ + c];
    }
    __shared__ float sp[NL_][256];
#pragma unroll
    for (int c = 0; c < NL_; ++c) sp[c][tid] = part[c];
    __syncthreads();
    for (int s = 128; s > 0; s >>= 1) {
        if (tid < s)
#pragma unroll
            for (int c = 0; c < NL_; ++c) sp[c][tid] += sp[c][tid + s];
        __syncthreads();
    }
    if (tid < NL_) out[((size_t)b * NS_ + k) * NL_ + tid] = sp[tid][0] + bc[tid];
}

// ---------------------------------------------------------------------------
extern "C" void kernel_launch(void* const* d_in, const int* in_sizes, int n_in,
                              void* d_out, int out_size, void* d_ws, size_t ws_size,
                              hipStream_t stream) {
    (void)in_sizes; (void)n_in; (void)out_size; (void)ws_size;
    const float* x      = (const float*)d_in[0];
    const int*   heads  = (const int*)d_in[1];
    const int*   tails  = (const int*)d_in[2];
    const int*   amask  = (const int*)d_in[3];
    const float* Wih_f  = (const float*)d_in[4];
    const float* Whh_f  = (const float*)d_in[5];
    const float* bih_f  = (const float*)d_in[6];
    const float* bhh_f  = (const float*)d_in[7];
    const float* Wih_b  = (const float*)d_in[8];
    const float* Whh_b  = (const float*)d_in[9];
    const float* bih_b  = (const float*)d_in[10];
    const float* bhh_b  = (const float*)d_in[11];
    const float* ln_g   = (const float*)d_in[12];
    const float* ln_b   = (const float*)d_in[13];
    const float* Wq     = (const float*)d_in[14];
    const float* bq     = (const float*)d_in[15];
    const float* Wk     = (const float*)d_in[16];
    const float* bk     = (const float*)d_in[17];
    const float* Wv     = (const float*)d_in[18];
    const float* bv     = (const float*)d_in[19];
    const float* Wo     = (const float*)d_in[20];
    const float* bo     = (const float*)d_in[21];
    const float* aln_g  = (const float*)d_in[22];
    const float* aln_b  = (const float*)d_in[23];
    const float* Wc     = (const float*)d_in[24];
    const float* bc     = (const float*)d_in[25];
    float* outp = (float*)d_out;

    // workspace layout (all offsets 256B-aligned; total ~766MB)
    char* ws = (char*)d_ws;
    const size_t OFF_XB16   = 0;                       // 67,108,864  x bf16
    const size_t OFF_WIH16  = OFF_XB16   + 67108864;   //  8,388,608  Wih bf16 [2]
    const size_t OFF_WHH16  = OFF_WIH16  + 8388608;    //  4,194,304  Whh bf16 [2]
    const size_t OFF_WQKVT  = OFF_WHH16  + 4194304;    //  6,291,456  WqT/WkT/WvT bf16
    const size_t OFF_WOT    = OFF_WQKVT  + 6291456;    //  2,097,152  WoT bf16
    const size_t OFF_XP     = OFF_WOT    + 2097152;    // 536,870,912 xp f32 [2]
    const size_t OFF_HHIST  = OFF_XP     + 536870912;  // 134,217,728 h_hist f32
    const size_t OFF_POOLED = OFF_HHIST  + 134217728;  //   1,048,576
    const size_t OFF_POOL16 = OFF_POOLED + 1048576;    //     524,288
    const size_t OFF_QKV    = OFF_POOL16 + 524288;     //   3,145,728
    const size_t OFF_CTX    = OFF_QKV    + 3145728;    //   1,048,576
    const size_t OFF_CTX16  = OFF_CTX    + 1048576;    //     524,288
    const size_t OFF_ATTN   = OFF_CTX16  + 524288;     //   1,048,576

    __bf16* xb16   = (__bf16*)(ws + OFF_XB16);
    __bf16* wih16  = (__bf16*)(ws + OFF_WIH16);
    __bf16* whh16  = (__bf16*)(ws + OFF_WHH16);
    __bf16* wqkvT  = (__bf16*)(ws + OFF_WQKVT);
    __bf16* woT    = (__bf16*)(ws + OFF_WOT);
    float*  xp     = (float*)(ws + OFF_XP);
    float*  hhist  = (float*)(ws + OFF_HHIST);
    float*  pooled = (float*)(ws + OFF_POOLED);
    __bf16* pool16 = (__bf16*)(ws + OFF_POOL16);
    float*  qkv    = (float*)(ws + OFF_QKV);
    float*  ctx    = (float*)(ws + OFF_CTX);
    __bf16* ctx16  = (__bf16*)(ws + OFF_CTX16);
    float*  attn   = (float*)(ws + OFF_ATTN);

    // 0) precision conversion / weight transposes
    k_cvt_bf16<<<4096, 256, 0, stream>>>(x, xb16, (size_t)B_ * S_ * H_);
    k_cvt_bf16<<<2048, 256, 0, stream>>>(Wih_f, wih16, (size_t)G4_ * H_);
    k_cvt_bf16<<<2048, 256, 0, stream>>>(Wih_b, wih16 + (size_t)G4_ * H_, (size_t)G4_ * H_);
    k_cvt_bf16<<<1024, 256, 0, stream>>>(Whh_f, whh16, (size_t)G4_ * HC_);
    k_cvt_bf16<<<1024, 256, 0, stream>>>(Whh_b, whh16 + (size_t)G4_ * HC_, (size_t)G4_ * HC_);
    k_cvtT_bf16<<<1024, 256, 0, stream>>>(Wq, wqkvT + 0 * (size_t)H_ * H_, H_, H_);
    k_cvtT_bf16<<<1024, 256, 0, stream>>>(Wk, wqkvT + 1 * (size_t)H_ * H_, H_, H_);
    k_cvtT_bf16<<<1024, 256, 0, stream>>>(Wv, wqkvT + 2 * (size_t)H_ * H_, H_, H_);
    k_cvtT_bf16<<<1024, 256, 0, stream>>>(Wo, woT, H_, H_);

    // 1) LSTM input projection: [32768x1024] x [1024x2048] x 2 dirs (WMMA bf16)
    //    4 M-tiles per wave -> grid.x = 32768/64
    k_input_proj<<<dim3(512, 16, 2), 256, 0, stream>>>(xb16, wih16, bih_f, bhh_f, bih_b, bhh_b, xp);

    // 2) sequential recurrence: one WGP per direction, 2048 steps
    k_lstm_scan<<<2, 1024, 0, stream>>>(xp, whh16, hhist);

    // 3) ragged span mean + layer norm
    k_span_pool<<<dim3(B_, NS_), 256, 0, stream>>>(hhist, heads, tails, ln_g, ln_b, pooled, pool16);

    // 4) Q/K/V projections (WMMA)
    k_qkv<<<dim3(B_, 8, 3), 256, 0, stream>>>(pool16, wqkvT, bq, bk, bv, qkv);

    // 5) softmax attention core (16x16 per head — negligible FLOPs)
    k_attn<<<dim3(B_, NH_), 256, 0, stream>>>(qkv, amask, ctx, ctx16);

    // 6) output projection (WMMA)
    k_oproj<<<dim3(B_, 8), 256, 0, stream>>>(ctx16, woT, bo, attn);

    // 7) residual + layer norm + classifier
    k_final<<<dim3(B_, NS_), 256, 0, stream>>>(attn, pooled, aln_g, aln_b, Wc, bc, outp);
}
